// GCNReg_3mlpdo_29703993819341
// MI455X (gfx1250) — matrix operand order repaired and checked by reading, not compiled, not run on hardware
//
#include <hip/hip_runtime.h>
#include <hip/hip_fp16.h>

#define HIDDIM 128

typedef float     v2f  __attribute__((ext_vector_type(2)));
typedef float     v8f  __attribute__((ext_vector_type(8)));
typedef _Float16  v16h __attribute__((ext_vector_type(16)));

// ---------------------------------------------------------------- utilities
__global__ void zero_f32(float* __restrict__ p, long n) {
    long i = (long)blockIdx.x * blockDim.x + threadIdx.x;
    if (i < n) p[i] = 0.0f;
}

// degrees: one thread per edge, two f32 atomics
__global__ void deg_kernel(const int* __restrict__ src, const int* __restrict__ dst,
                           float* __restrict__ dsrc, float* __restrict__ ddst, int E) {
    int e = blockIdx.x * blockDim.x + threadIdx.x;
    if (e < E) {
        atomicAdd(dsrc + src[e], 1.0f);
        atomicAdd(ddst + dst[e], 1.0f);
    }
}

// deg -> rsqrt(max(deg,1)) in place (dsrc/ddst are contiguous; n = 2*N)
__global__ void isqrt_kernel(float* __restrict__ d, long n) {
    long i = (long)blockIdx.x * blockDim.x + threadIdx.x;
    if (i < n) d[i] = rsqrtf(fmaxf(d[i], 1.0f));
}

// ---------------------------------------------------------------- WMMA GEMM
// out[r, :] = (A[r, :] @ W) * rowscale[r]     A: [N,128], W: [128,128]
// block = 256 threads = 8 waves; block handles a 16-row tile; wave w handles
// columns 16w .. 16w+15. fp32 WMMA 16x16x4 (exact) with f16 fallback.
__global__ __launch_bounds__(256) void gcn_gemm_rowscale(
    const float* __restrict__ A, const float* __restrict__ W,
    const float* __restrict__ rowscale, float* __restrict__ out)
{
    __shared__ float Al[16 * HIDDIM];   // 8 KB A tile
    __shared__ float rs[16];

    const int rowbase = blockIdx.x * 16;
    {   // cooperative tile load, float4-coalesced
        const float4* Ag = (const float4*)(A + (long)rowbase * HIDDIM);
        float4* Ad = (float4*)Al;
        for (int i = threadIdx.x; i < 16 * (HIDDIM / 4); i += 256) Ad[i] = Ag[i];
        if (threadIdx.x < 16) rs[threadIdx.x] = rowscale[rowbase + threadIdx.x];
    }
    __syncthreads();

    const int lane = threadIdx.x & 31;
    const int wave = threadIdx.x >> 5;
    const int m    = lane & 15;     // row (A frag) / col (B,D frags)
    const int h    = lane >> 4;     // lane half selects K group
    const int col  = wave * 16 + m;

    v8f c = {};
#if __has_builtin(__builtin_amdgcn_wmma_f32_16x16x4_f32)
    // native fp32 matrix path: A frag v = A[m][k+2h+v], B frag v = W[k+2h+v][col]
    for (int k = 0; k < HIDDIM; k += 4) {
        v2f a = *(const v2f*)(Al + m * HIDDIM + k + 2 * h);   // 8B-aligned LDS read
        v2f b;
        b.x = W[(k + 2 * h)     * HIDDIM + col];
        b.y = W[(k + 2 * h + 1) * HIDDIM + col];
        c = __builtin_amdgcn_wmma_f32_16x16x4_f32(false, a, false, b,
                                                  (short)0, c, false, false);
    }
#else
    // fallback: f16 inputs, f32 accumulate, 16x16x32 (codegen-confirmed builtin)
    for (int k = 0; k < HIDDIM; k += 32) {
        v16h a, b;
#pragma unroll
        for (int j = 0; j < 8; ++j) {
            int ka = k + ((j < 4) ? (8 * h + 2 * j) : (16 + 8 * h + 2 * (j - 4)));
            a[2 * j]     = (_Float16)Al[m * HIDDIM + ka];
            a[2 * j + 1] = (_Float16)Al[m * HIDDIM + ka + 1];
            int kb = k + ((j < 4) ? (8 * h + 2 * j) : (16 + 8 * h + 2 * (j - 4)));
            b[2 * j]     = (_Float16)W[kb * HIDDIM + col];
            b[2 * j + 1] = (_Float16)W[(kb + 1) * HIDDIM + col];
        }
        c = __builtin_amdgcn_wmma_f32_16x16x32_f16(false, a, false, b,
                                                   (short)0, c, false, false);
    }
#endif
    // D layout: VGPR r -> row r + 8h, col m; fold src_isqrt row scaling here
#pragma unroll
    for (int r = 0; r < 8; ++r) {
        int row = r + 8 * h;
        out[(long)(rowbase + row) * HIDDIM + col] = c[r] * rs[row];
    }
}

// ---------------------------------------------------------------- edge scatter
// agg[dst[e], :] += msg[src[e], :]   one wave per edge, float4 gather,
// 4 f32 atomics per lane (addresses stay in 192MB L2 -> L2-rate atomics)
__global__ __launch_bounds__(256) void scatter_kernel(
    const float* __restrict__ msg, const int* __restrict__ src,
    const int* __restrict__ dst, float* __restrict__ agg, int E)
{
    int e = blockIdx.x * 8 + (threadIdx.x >> 5);
    if (e >= E) return;
    int lane = threadIdx.x & 31;
    int s = src[e], d = dst[e];
    float4 v = ((const float4*)(msg + (long)s * HIDDIM))[lane];
    float* dp = agg + (long)d * HIDDIM + lane * 4;
    atomicAdd(dp + 0, v.x);
    atomicAdd(dp + 1, v.y);
    atomicAdd(dp + 2, v.z);
    atomicAdd(dp + 3, v.w);
}

// h = relu(agg * dst_isqrt[node] + bias)   (vectorized, one float4 per thread)
__global__ void relunorm_kernel(const float* __restrict__ agg,
                                const float* __restrict__ disqrt,
                                const float* __restrict__ bias,
                                float* __restrict__ h, int n)
{
    long i = (long)blockIdx.x * blockDim.x + threadIdx.x;   // over n*32 float4s
    if (i >= (long)n * (HIDDIM / 4)) return;
    int node = (int)(i >> 5);
    int f4   = (int)(i & 31);
    float  s = disqrt[node];
    float4 v = ((const float4*)agg)[i];
    float4 b = ((const float4*)bias)[f4];
    float4 o;
    o.x = fmaxf(v.x * s + b.x, 0.0f);
    o.y = fmaxf(v.y * s + b.y, 0.0f);
    o.z = fmaxf(v.z * s + b.z, 0.0f);
    o.w = fmaxf(v.w * s + b.w, 0.0f);
    ((float4*)h)[i] = o;
}

// per-graph feature sums + node counts
__global__ void pool_kernel(const float* __restrict__ h, const int* __restrict__ gid,
                            float* __restrict__ gsum, float* __restrict__ gcnt, int n)
{
    long i = (long)blockIdx.x * blockDim.x + threadIdx.x;   // over n*32 float4s
    if (i >= (long)n * (HIDDIM / 4)) return;
    int node = (int)(i >> 5);
    int f4   = (int)(i & 31);
    int g    = gid[node];
    float4 v = ((const float4*)h)[i];
    float* gp = gsum + (long)g * HIDDIM + f4 * 4;
    atomicAdd(gp + 0, v.x);
    atomicAdd(gp + 1, v.y);
    atomicAdd(gp + 2, v.z);
    atomicAdd(gp + 3, v.w);
    if (f4 == 0) atomicAdd(gcnt + g, 1.0f);
}

// mean + 4-layer MLP head, single block, all intermediates in LDS (60 KB)
__global__ __launch_bounds__(256) void head_kernel(
    const float* __restrict__ gsum, const float* __restrict__ gcnt,
    const float* __restrict__ Wc1, const float* __restrict__ bc1,
    const float* __restrict__ Wc2, const float* __restrict__ bc2,
    const float* __restrict__ Wc3, const float* __restrict__ bc3,
    const float* __restrict__ Wc4, const float* __restrict__ bc4,
    float* __restrict__ out)
{
    __shared__ float hg[64 * 128];
    __shared__ float o1[64 * 64];
    __shared__ float o2[64 * 32];
    __shared__ float o3[64 * 16];
    int tid = threadIdx.x;
    for (int i = tid; i < 64 * 128; i += 256)
        hg[i] = gsum[i] / fmaxf(gcnt[i >> 7], 1.0f);
    __syncthreads();
    for (int i = tid; i < 64 * 64; i += 256) {
        int g = i >> 6, j = i & 63;
        float acc = bc1[j];
        for (int k = 0; k < 128; ++k) acc += hg[g * 128 + k] * Wc1[k * 64 + j];
        o1[i] = fmaxf(acc, 0.0f);
    }
    __syncthreads();
    for (int i = tid; i < 64 * 32; i += 256) {
        int g = i >> 5, j = i & 31;
        float acc = bc2[j];
        for (int k = 0; k < 64; ++k) acc += o1[g * 64 + k] * Wc2[k * 32 + j];
        o2[i] = fmaxf(acc, 0.0f);
    }
    __syncthreads();
    for (int i = tid; i < 64 * 16; i += 256) {
        int g = i >> 4, j = i & 15;
        float acc = bc3[j];
        for (int k = 0; k < 32; ++k) acc += o2[g * 32 + k] * Wc3[k * 16 + j];
        o3[i] = fmaxf(acc, 0.0f);
    }
    __syncthreads();
    if (tid < 64) {
        float acc = bc4[0];
        for (int k = 0; k < 16; ++k) acc += o3[tid * 16 + k] * Wc4[k];
        out[tid] = acc;
    }
}

// ---------------------------------------------------------------- launcher
extern "C" void kernel_launch(void* const* d_in, const int* in_sizes, int n_in,
                              void* d_out, int out_size, void* d_ws, size_t ws_size,
                              hipStream_t stream)
{
    const float* x   = (const float*)d_in[0];
    const int*   src = (const int*)d_in[1];
    const int*   dst = (const int*)d_in[2];
    const int*   gid = (const int*)d_in[3];
    // d_in[4] = num_graphs (fixed 64 in this problem)
    const float* W1  = (const float*)d_in[5];
    const float* b1  = (const float*)d_in[6];
    const float* W2  = (const float*)d_in[7];
    const float* b2  = (const float*)d_in[8];
    const float* Wc1 = (const float*)d_in[9];
    const float* bc1 = (const float*)d_in[10];
    const float* Wc2 = (const float*)d_in[11];
    const float* bc2 = (const float*)d_in[12];
    const float* Wc3 = (const float*)d_in[13];
    const float* bc3 = (const float*)d_in[14];
    const float* Wc4 = (const float*)d_in[15];
    const float* bc4 = (const float*)d_in[16];

    const int  N  = in_sizes[3];          // 50000 (divisible by 16)
    const int  E  = in_sizes[1];          // 1.6M
    const long NF = (long)N * HIDDIM;

    float* ws   = (float*)d_ws;
    float* bufA = ws;                     // N*128
    float* bufB = ws + NF;                // N*128
    float* dsrc = ws + 2 * NF;            // N
    float* ddst = dsrc + N;               // N
    float* gsum = ddst + N;               // 64*128
    float* gcnt = gsum + 64 * HIDDIM;     // 64

    auto cdiv = [](long a, long b) { return (int)((a + b - 1) / b); };

    // zero degrees + pooling accumulators (contiguous tail region)
    const long tail = 2L * N + 64 * HIDDIM + 64;
    zero_f32<<<cdiv(tail, 256), 256, 0, stream>>>(dsrc, tail);

    deg_kernel<<<cdiv(E, 256), 256, 0, stream>>>(src, dst, dsrc, ddst, E);
    isqrt_kernel<<<cdiv(2L * N, 256), 256, 0, stream>>>(dsrc, 2L * N);

    // ---- layer 1:  msg = (x @ W1) * src_isqrt ; agg = scatter ; h1 = relu(norm)
    gcn_gemm_rowscale<<<N / 16, 256, 0, stream>>>(x, W1, dsrc, bufA);
    zero_f32<<<cdiv(NF, 256), 256, 0, stream>>>(bufB, NF);
    scatter_kernel<<<cdiv(E, 8), 256, 0, stream>>>(bufA, src, dst, bufB, E);
    relunorm_kernel<<<cdiv((long)N * 32, 256), 256, 0, stream>>>(bufB, ddst, b1, bufA, N);

    // ---- layer 2
    gcn_gemm_rowscale<<<N / 16, 256, 0, stream>>>(bufA, W2, dsrc, bufB);
    zero_f32<<<cdiv(NF, 256), 256, 0, stream>>>(bufA, NF);
    scatter_kernel<<<cdiv(E, 8), 256, 0, stream>>>(bufB, src, dst, bufA, E);
    relunorm_kernel<<<cdiv((long)N * 32, 256), 256, 0, stream>>>(bufA, ddst, b2, bufB, N);

    // ---- mean-pool + MLP head
    pool_kernel<<<cdiv((long)N * 32, 256), 256, 0, stream>>>(bufB, gid, gsum, gcnt, N);
    head_kernel<<<1, 256, 0, stream>>>(gsum, gcnt, Wc1, bc1, Wc2, bc2,
                                       Wc3, bc3, Wc4, bc4, (float*)d_out);
}
